// AttentionConv1dCogVideoXAttnProcessor2_0_58875411694117
// MI455X (gfx1250) — compile-verified, matrix operand order checked by reading, not run on hardware
//
#include <hip/hip_runtime.h>
#include <hip/hip_bf16.h>
#include <math.h>

// ---------------- problem constants ----------------
#define DIM_   1920
#define HEADS_ 30
#define HD_    64
#define RANK_  128
#define T_     226
#define SV_    2048           // F*H*W
#define S_     2274           // T + SV
#define SP_    2304           // S padded to 18*128 (multiple of 128 and 16)

typedef __attribute__((ext_vector_type(16))) _Float16 v16h;
typedef __attribute__((ext_vector_type(8)))  _Float16 v8h;
typedef __attribute__((ext_vector_type(8)))  float    v8f;

union AFrag { v16h v; v8h h[2]; };

// ---- CDNA5 async global->LDS (ASYNCcnt-tracked), via inline asm ----
__device__ __forceinline__ void async_ldsb128(unsigned ldsByteOff, const void* g) {
    asm volatile("global_load_async_to_lds_b128 %0, %1, off"
                 :: "v"(ldsByteOff), "v"((unsigned long long)(uintptr_t)g)
                 : "memory");
}
__device__ __forceinline__ void wait_async0() {           // drain all
    asm volatile("s_wait_asynccnt 0x0" ::: "memory");
}
__device__ __forceinline__ void wait_async4() {           // oldest buffer landed
    asm volatile("s_wait_asynccnt 0x4" ::: "memory");
}

// =====================================================================
// Elementwise prep kernels
// =====================================================================
__global__ void build_hs16_kernel(const float* __restrict__ enc,
                                  const float* __restrict__ hid,
                                  _Float16* __restrict__ dst) {
    size_t idx = (size_t)blockIdx.x * blockDim.x + threadIdx.x;
    size_t total = (size_t)SP_ * DIM_;
    if (idx >= total) return;
    int m = (int)(idx / DIM_);
    int c = (int)(idx % DIM_);
    float v = 0.f;
    if (m < T_)       v = enc[(size_t)m * DIM_ + c];
    else if (m < S_)  v = hid[(size_t)(m - T_) * DIM_ + c];
    dst[idx] = (_Float16)v;
}

__global__ void convert_f16_kernel(const float* __restrict__ src,
                                   _Float16* __restrict__ dst, size_t n) {
    size_t idx = (size_t)blockIdx.x * blockDim.x + threadIdx.x;
    if (idx < n) dst[idx] = (_Float16)src[idx];
}

// conv1_w (RANK, DIM, 3) -> w1t[tap][o][i]
__global__ void repack_conv1_kernel(const float* __restrict__ src,
                                    _Float16* __restrict__ dst) {
    size_t idx = (size_t)blockIdx.x * blockDim.x + threadIdx.x;
    size_t per = (size_t)RANK_ * DIM_;
    if (idx >= 3 * per) return;
    int tap = (int)(idx / per);
    size_t rem = idx - (size_t)tap * per;      // o*DIM + i
    dst[idx] = (_Float16)src[rem * 3 + tap];
}

// conv2_w (DIM, RANK, 3) -> w2t[tap][d][r]
__global__ void repack_conv2_kernel(const float* __restrict__ src,
                                    _Float16* __restrict__ dst) {
    size_t idx = (size_t)blockIdx.x * blockDim.x + threadIdx.x;
    size_t per = (size_t)DIM_ * RANK_;
    if (idx >= 3 * per) return;
    int tap = (int)(idx / per);
    size_t rem = idx - (size_t)tap * per;      // d*RANK + r
    dst[idx] = (_Float16)src[rem * 3 + tap];
}

// attn + conv_branch -> f16 projection input (padded rows = 0)
__global__ void combine_kernel(const float* __restrict__ attn,
                               const float* __restrict__ cbr,
                               _Float16* __restrict__ dst) {
    size_t idx = (size_t)blockIdx.x * blockDim.x + threadIdx.x;
    size_t total = (size_t)SP_ * DIM_;
    if (idx >= total) return;
    int m = (int)(idx / DIM_);
    int c = (int)(idx % DIM_);
    float v = 0.f;
    if (m < S_) {
        v = attn[idx];
        if (m >= T_) v += cbr[(size_t)(m - T_) * DIM_ + c];
    }
    dst[idx] = (_Float16)v;
}

// =====================================================================
// Tiled WMMA GEMM:  C[M,N] = sum_taps A_shift[M,K] * W_tap[N,K]^T (+bias)
// BM=128 BN=128 BK=32, 8 waves (4 M x 2 N), each wave -> 32x64 (8 WMMA/step).
// ASYNC=true : triple-buffered LDS filled by global_load_async_to_lds_b128,
//              partial s_wait_asynccnt (2-step prefetch distance), one barrier
//              per K-step (rows must all be valid; ntaps==1).
// ASYNC=false: synchronous fill with row-shift bounds checks (conv taps).
// mode: 0 = f32 store, 1 = GELU + f16 store, 2 = f32 permuted store (final proj)
// =====================================================================
#define LDS_STRIDE 40   // f16 elements; 80B row stride keeps 16B alignment

template<bool ASYNC>
__launch_bounds__(256)
__global__ void gemm_wmma_kernel(const _Float16* __restrict__ A, int lda, int M,
                                 const _Float16* __restrict__ Wm, int ldw,
                                 int tapStrideW, int ntaps, int rowShift,
                                 const float* __restrict__ bias,
                                 void* __restrict__ Cout, int ldc,
                                 int K, int mode) {
    __shared__ __align__(16) _Float16 lA[3][128 * LDS_STRIDE];
    __shared__ __align__(16) _Float16 lB[3][128 * LDS_STRIDE];

    const int tid  = threadIdx.x;
    const int wave = tid >> 5;
    const int lane = tid & 31;
    const int wm   = wave >> 1;          // 0..3 -> 32-row slice
    const int wn   = wave & 1;           // 0..1 -> 64-col slice
    const int half = lane >> 4;
    const int l16  = lane & 15;
    const int mBase = blockIdx.y * 128;
    const int nBase = blockIdx.x * 128;

    const int ksteps = K >> 5;
    const int nsteps = ntaps * ksteps;

    // per-thread staging slot: row r, 16B chunks at cg and cg+8 (f16 elems)
    const int r  = tid >> 1;
    const int cg = (tid & 1) * 16;

    v8f acc[2][4];
#pragma unroll
    for (int i = 0; i < 2; ++i)
#pragma unroll
        for (int j = 0; j < 4; ++j)
#pragma unroll
            for (int e = 0; e < 8; ++e) acc[i][j][e] = 0.f;

    auto issue_async = [&](int step, int buf) {
        const int k0 = (step % ksteps) << 5;           // ntaps==1 in async path
        const _Float16* aSrc = A  + (size_t)(mBase + r) * lda + k0 + cg;
        const _Float16* bSrc = Wm + (size_t)(nBase + r) * ldw + k0 + cg;
        unsigned la = (unsigned)(uintptr_t)(&lA[buf][r * LDS_STRIDE + cg]);
        unsigned lb = (unsigned)(uintptr_t)(&lB[buf][r * LDS_STRIDE + cg]);
        async_ldsb128(la,      aSrc);
        async_ldsb128(la + 16, aSrc + 8);
        async_ldsb128(lb,      bSrc);
        async_ldsb128(lb + 16, bSrc + 8);
    };

    auto compute = [&](int buf) {
        AFrag a0, a1;
        const _Float16* ap0 = lA[buf] + (wm * 32 + l16) * LDS_STRIDE;
        const _Float16* ap1 = ap0 + 16 * LDS_STRIDE;
        a0.h[0] = *(const v8h*)(ap0 + 8 * half);
        a0.h[1] = *(const v8h*)(ap0 + 16 + 8 * half);
        a1.h[0] = *(const v8h*)(ap1 + 8 * half);
        a1.h[1] = *(const v8h*)(ap1 + 16 + 8 * half);
#pragma unroll
        for (int nt = 0; nt < 4; ++nt) {
            AFrag b;
            const _Float16* bp = lB[buf] + (wn * 64 + nt * 16 + l16) * LDS_STRIDE;
            b.h[0] = *(const v8h*)(bp + 8 * half);
            b.h[1] = *(const v8h*)(bp + 16 + 8 * half);
            acc[0][nt] = __builtin_amdgcn_wmma_f32_16x16x32_f16(
                false, a0.v, false, b.v, (short)0, acc[0][nt], false, false);
            acc[1][nt] = __builtin_amdgcn_wmma_f32_16x16x32_f16(
                false, a1.v, false, b.v, (short)0, acc[1][nt], false, false);
        }
    };

    if (ASYNC) {
        issue_async(0, 0);
        if (nsteps > 1) issue_async(1, 1);
        for (int step = 0; step < nsteps; ++step) {
            const int buf = step % 3;
            if (step + 1 < nsteps) wait_async4();   // oldest 4 (this buf) landed
            else                   wait_async0();
            __syncthreads();                        // all waves' fills visible
            if (step + 2 < nsteps) issue_async(step + 2, (step + 2) % 3);
            compute(buf);
        }
    } else {
        for (int step = 0; step < nsteps; ++step) {
            const int tap = step / ksteps;
            const int k0  = (step - tap * ksteps) << 5;
            const int shift = (tap - (ntaps >> 1)) * rowShift;   // -R,0,+R
            const _Float16* Wt = Wm + (size_t)tap * tapStrideW;
            // A tile 128x32 with bounds (conv time-shift zero padding)
            {
                int gm = mBase + r + shift;
                v8h v0, v1;
#pragma unroll
                for (int e = 0; e < 8; ++e) { v0[e] = (_Float16)0.f; v1[e] = (_Float16)0.f; }
                if (gm >= 0 && gm < M) {
                    const _Float16* src = A + (size_t)gm * lda + k0 + cg;
                    v0 = *(const v8h*)(src);
                    v1 = *(const v8h*)(src + 8);
                }
                *(v8h*)(lA[0] + r * LDS_STRIDE + cg)     = v0;
                *(v8h*)(lA[0] + r * LDS_STRIDE + cg + 8) = v1;
            }
            // B tile 128x32
            {
                const _Float16* src = Wt + (size_t)(nBase + r) * ldw + k0 + cg;
                *(v8h*)(lB[0] + r * LDS_STRIDE + cg)     = *(const v8h*)(src);
                *(v8h*)(lB[0] + r * LDS_STRIDE + cg + 8) = *(const v8h*)(src + 8);
            }
            __syncthreads();
            compute(0);
            __syncthreads();
        }
    }

    // ---- epilogue: C layout row = r + 8*half, col = lane%16 ----
#pragma unroll
    for (int mi = 0; mi < 2; ++mi) {
        const int mRow0 = mBase + wm * 32 + mi * 16 + 8 * half;
#pragma unroll
        for (int nt = 0; nt < 4; ++nt) {
            int col = nBase + wn * 64 + nt * 16 + l16;
            float bv = bias ? bias[col] : 0.f;
#pragma unroll
            for (int rr = 0; rr < 8; ++rr) {
                float val = acc[mi][nt][rr] + bv;
                int m = mRow0 + rr;
                if (mode == 0) {
                    ((float*)Cout)[(size_t)m * ldc + col] = val;
                } else if (mode == 1) {
                    float g = 0.5f * val * (1.f + erff(val * 0.70710678118f));
                    ((_Float16*)Cout)[(size_t)m * ldc + col] = (_Float16)g;
                } else {
                    if (m < S_) {
                        int dm = (m >= T_) ? (m - T_) : (SV_ + m);
                        ((float*)Cout)[(size_t)dm * ldc + col] = val;
                    }
                }
            }
        }
    }
}

// =====================================================================
// Per-(token, head) LayerNorm + RoPE; emit q/k f16 [head][s][64] and
// v transposed f16 [head][d][s].  One wave per (s, head); lane owns
// dims {lane, lane+32}.  q is pre-scaled by 1/sqrt(HD).
// =====================================================================
__launch_bounds__(32)
__global__ void ln_rope_kernel(const float* __restrict__ q32,
                               const float* __restrict__ k32,
                               const float* __restrict__ v32,
                               const float* __restrict__ cosb,
                               const float* __restrict__ sinb,
                               const float* __restrict__ nqw, const float* __restrict__ nqb,
                               const float* __restrict__ nkw, const float* __restrict__ nkb,
                               _Float16* __restrict__ q16,
                               _Float16* __restrict__ k16,
                               _Float16* __restrict__ vt16) {
    const int blk  = blockIdx.x;
    const int s    = blk / HEADS_;
    const int head = blk % HEADS_;
    const int lane = threadIdx.x;

    _Float16* qd = q16 + ((size_t)head * SP_ + s) * HD_;
    _Float16* kd = k16 + ((size_t)head * SP_ + s) * HD_;
    _Float16* v0 = vt16 + ((size_t)head * HD_ + lane) * SP_ + s;
    _Float16* v1 = vt16 + ((size_t)head * HD_ + lane + 32) * SP_ + s;

    if (s >= S_) {
        qd[lane] = (_Float16)0.f; qd[lane + 32] = (_Float16)0.f;
        kd[lane] = (_Float16)0.f; kd[lane + 32] = (_Float16)0.f;
        *v0 = (_Float16)0.f; *v1 = (_Float16)0.f;
        return;
    }
    const size_t base = (size_t)s * DIM_ + head * HD_;
    *v0 = (_Float16)v32[base + lane];
    *v1 = (_Float16)v32[base + lane + 32];

    const bool rope = (s >= T_);
    const float* cr = rope ? (cosb + (size_t)(s - T_) * HD_) : nullptr;
    const float* sr = rope ? (sinb + (size_t)(s - T_) * HD_) : nullptr;

#pragma unroll
    for (int which = 0; which < 2; ++which) {
        const float* src = which ? k32 : q32;
        const float* nw  = which ? nkw : nqw;
        const float* nb  = which ? nkb : nqb;
        _Float16*    dst = which ? kd  : qd;

        float x0 = src[base + lane], x1 = src[base + lane + 32];
        float sum = x0 + x1;
#pragma unroll
        for (int off = 16; off >= 1; off >>= 1) sum += __shfl_xor(sum, off, 32);
        float mu = sum * (1.f / 64.f);
        float d0 = x0 - mu, d1 = x1 - mu;
        float vs = d0 * d0 + d1 * d1;
#pragma unroll
        for (int off = 16; off >= 1; off >>= 1) vs += __shfl_xor(vs, off, 32);
        float rs = rsqrtf(vs * (1.f / 64.f) + 1e-6f);
        float y0 = d0 * rs * nw[lane] + nb[lane];
        float y1 = d1 * rs * nw[lane + 32] + nb[lane + 32];

        if (rope) {
            float p0 = __shfl_xor(y0, 1, 32);
            float p1 = __shfl_xor(y1, 1, 32);
            float r0 = (lane & 1) ? p0 : -p0;   // even d: -x[d+1], odd d: +x[d-1]
            float r1 = (lane & 1) ? p1 : -p1;
            y0 = y0 * cr[lane]      + r0 * sr[lane];
            y1 = y1 * cr[lane + 32] + r1 * sr[lane + 32];
        }
        if (which == 0) { y0 *= 0.125f; y1 *= 0.125f; }   // fold 1/sqrt(64) into q
        dst[lane]      = (_Float16)y0;
        dst[lane + 32] = (_Float16)y1;
    }
}

// =====================================================================
// Flash-style attention: one wave = 16 query rows of one head.
// Per 32-key step: 4 WMMA for scores, online softmax (fully unrolled,
// register-resident), P through LDS, 4 WMMA for P·V.  Next K/V tiles
// prefetched (global_prefetch_b8).
// =====================================================================
__launch_bounds__(128)
__global__ void attn_wmma_kernel(const _Float16* __restrict__ q16,
                                 const _Float16* __restrict__ k16,
                                 const _Float16* __restrict__ vt16,
                                 float* __restrict__ attnOut) {
    __shared__ __align__(16) _Float16 lP[4][16 * 32];

    const int wave = threadIdx.x >> 5;
    const int lane = threadIdx.x & 31;
    const int half = lane >> 4;
    const int l16  = lane & 15;
    const int head = blockIdx.y;
    const int qt   = blockIdx.x * 4 + wave;       // 0..143 (grid.x = 36)

    const _Float16* qh = q16 + ((size_t)head * SP_ + qt * 16) * HD_;
    const _Float16* kh = k16 + (size_t)head * SP_ * HD_;
    const _Float16* vh = vt16 + (size_t)head * HD_ * SP_;

    AFrag aq0, aq1;
    {
        const _Float16* qp = qh + (size_t)l16 * HD_;
        aq0.h[0] = *(const v8h*)(qp + 8 * half);
        aq0.h[1] = *(const v8h*)(qp + 16 + 8 * half);
        aq1.h[0] = *(const v8h*)(qp + 32 + 8 * half);
        aq1.h[1] = *(const v8h*)(qp + 48 + 8 * half);
    }

    v8f o[4];
#pragma unroll
    for (int i = 0; i < 4; ++i)
#pragma unroll
        for (int e = 0; e < 8; ++e) o[i][e] = 0.f;
    float mrow[8], lrow[8];
#pragma unroll
    for (int r = 0; r < 8; ++r) { mrow[r] = -3e38f; lrow[r] = 0.f; }

    for (int t = 0; t < SP_ / 32; ++t) {          // 72 key-pair steps
        if (t + 1 < SP_ / 32) {                   // prefetch next K/V tiles
            __builtin_prefetch(kh + (size_t)(t * 32 + 32 + l16) * HD_, 0, 1);
            __builtin_prefetch(vh + (size_t)l16 * SP_ + t * 32 + 32, 0, 1);
        }
        v8f sc[2];
#pragma unroll
        for (int jj = 0; jj < 2; ++jj) {
#pragma unroll
            for (int e = 0; e < 8; ++e) sc[jj][e] = 0.f;
            const _Float16* kp = kh + (size_t)(t * 32 + jj * 16 + l16) * HD_;
            AFrag b0, b1;
            b0.h[0] = *(const v8h*)(kp + 8 * half);
            b0.h[1] = *(const v8h*)(kp + 16 + 8 * half);
            b1.h[0] = *(const v8h*)(kp + 32 + 8 * half);
            b1.h[1] = *(const v8h*)(kp + 48 + 8 * half);
            sc[jj] = __builtin_amdgcn_wmma_f32_16x16x32_f16(
                false, aq0.v, false, b0.v, (short)0, sc[jj], false, false);
            sc[jj] = __builtin_amdgcn_wmma_f32_16x16x32_f16(
                false, aq1.v, false, b1.v, (short)0, sc[jj], false, false);
        }
        if (t * 32 + l16 >= S_) {
#pragma unroll
            for (int r = 0; r < 8; ++r) sc[0][r] = -3e38f;
        }
        if (t * 32 + 16 + l16 >= S_) {
#pragma unroll
            for (int r = 0; r < 8; ++r) sc[1][r] = -3e38f;
        }

#pragma unroll
        for (int r = 0; r < 8; ++r) {
            float mx = fmaxf(sc[0][r], sc[1][r]);
#pragma unroll
            for (int off = 8; off >= 1; off >>= 1) mx = fmaxf(mx, __shfl_xor(mx, off, 32));
            float mnew = fmaxf(mrow[r], mx);
            float corr = __expf(mrow[r] - mnew);
            float p0 = __expf(sc[0][r] - mnew);
            float p1 = __expf(sc[1][r] - mnew);
            float ps = p0 + p1;
#pragma unroll
            for (int off = 8; off >= 1; off >>= 1) ps += __shfl_xor(ps, off, 32);
            lrow[r] = lrow[r] * corr + ps;
            mrow[r] = mnew;
#pragma unroll
            for (int nt = 0; nt < 4; ++nt) o[nt][r] *= corr;
            lP[wave][(r + 8 * half) * 32 + l16]      = (_Float16)p0;
            lP[wave][(r + 8 * half) * 32 + 16 + l16] = (_Float16)p1;
        }
        AFrag ap;
        const _Float16* pp = &lP[wave][l16 * 32];
        ap.h[0] = *(const v8h*)(pp + 8 * half);
        ap.h[1] = *(const v8h*)(pp + 16 + 8 * half);
#pragma unroll
        for (int nt = 0; nt < 4; ++nt) {
            const _Float16* vp = vh + (size_t)(nt * 16 + l16) * SP_ + t * 32;
            AFrag bv;
            bv.h[0] = *(const v8h*)(vp + 8 * half);
            bv.h[1] = *(const v8h*)(vp + 16 + 8 * half);
            o[nt] = __builtin_amdgcn_wmma_f32_16x16x32_f16(
                false, ap.v, false, bv.v, (short)0, o[nt], false, false);
        }
    }

#pragma unroll
    for (int nt = 0; nt < 4; ++nt)
#pragma unroll
        for (int r = 0; r < 8; ++r) {
            int m = qt * 16 + r + 8 * half;
            if (m < S_)
                attnOut[(size_t)m * DIM_ + head * HD_ + nt * 16 + l16] =
                    o[nt][r] / lrow[r];
        }
}

// =====================================================================
// Host-side orchestration
// =====================================================================
extern "C" void kernel_launch(void* const* d_in, const int* in_sizes, int n_in,
                              void* d_out, int out_size, void* d_ws, size_t ws_size,
                              hipStream_t stream) {
    const float* hid   = (const float*)d_in[0];
    const float* enc   = (const float*)d_in[1];
    const float* cosb  = (const float*)d_in[2];
    const float* sinb  = (const float*)d_in[3];
    const float* wq    = (const float*)d_in[4];
    const float* bq    = (const float*)d_in[5];
    const float* wk    = (const float*)d_in[6];
    const float* bk    = (const float*)d_in[7];
    const float* wv    = (const float*)d_in[8];
    const float* bv    = (const float*)d_in[9];
    const float* nqw   = (const float*)d_in[10];
    const float* nqb   = (const float*)d_in[11];
    const float* nkw   = (const float*)d_in[12];
    const float* nkb   = (const float*)d_in[13];
    const float* c1w   = (const float*)d_in[14];
    const float* c2w   = (const float*)d_in[15];
    const float* wo    = (const float*)d_in[16];
    const float* bo    = (const float*)d_in[17];
    float* out = (float*)d_out;

    char* ws = (char*)d_ws;
    size_t off = 0;
    auto alloc = [&](size_t bytes) -> void* {
        void* p = ws + off;
        off = (off + bytes + 255) & ~(size_t)255;
        return p;
    };
    _Float16* hs16   = (_Float16*)alloc((size_t)SP_ * DIM_ * 2);
    _Float16* wq16   = (_Float16*)alloc((size_t)DIM_ * DIM_ * 2);
    _Float16* wk16   = (_Float16*)alloc((size_t)DIM_ * DIM_ * 2);
    _Float16* wv16   = (_Float16*)alloc((size_t)DIM_ * DIM_ * 2);
    _Float16* wo16   = (_Float16*)alloc((size_t)DIM_ * DIM_ * 2);
    float*    q32    = (float*)alloc((size_t)SP_ * DIM_ * 4);
    float*    k32    = (float*)alloc((size_t)SP_ * DIM_ * 4);
    float*    v32    = (float*)alloc((size_t)SP_ * DIM_ * 4);
    _Float16* q16    = (_Float16*)alloc((size_t)HEADS_ * SP_ * HD_ * 2);
    _Float16* k16    = (_Float16*)alloc((size_t)HEADS_ * SP_ * HD_ * 2);
    _Float16* vt16   = (_Float16*)alloc((size_t)HEADS_ * HD_ * SP_ * 2);
    float*    attn32 = (float*)alloc((size_t)SP_ * DIM_ * 4);
    _Float16* w1t16  = (_Float16*)alloc((size_t)3 * RANK_ * DIM_ * 2);
    _Float16* c1o16  = (_Float16*)alloc((size_t)SV_ * RANK_ * 2);
    _Float16* w2t16  = (_Float16*)alloc((size_t)3 * DIM_ * RANK_ * 2);
    float*    cbr32  = (float*)alloc((size_t)SV_ * DIM_ * 4);
    _Float16* pin16  = (_Float16*)alloc((size_t)SP_ * DIM_ * 2);

    const size_t nHS = (size_t)SP_ * DIM_;
    const size_t nW  = (size_t)DIM_ * DIM_;

    build_hs16_kernel<<<(nHS + 255) / 256, 256, 0, stream>>>(enc, hid, hs16);
    convert_f16_kernel<<<(nW + 255) / 256, 256, 0, stream>>>(wq, wq16, nW);
    convert_f16_kernel<<<(nW + 255) / 256, 256, 0, stream>>>(wk, wk16, nW);
    convert_f16_kernel<<<(nW + 255) / 256, 256, 0, stream>>>(wv, wv16, nW);
    convert_f16_kernel<<<(nW + 255) / 256, 256, 0, stream>>>(wo, wo16, nW);
    repack_conv1_kernel<<<((size_t)3 * RANK_ * DIM_ + 255) / 256, 256, 0, stream>>>(c1w, w1t16);
    repack_conv2_kernel<<<((size_t)3 * DIM_ * RANK_ + 255) / 256, 256, 0, stream>>>(c2w, w2t16);

    // QKV projections: M=2304, N=1920, K=1920 (async triple-buffered path)
    dim3 gProj(DIM_ / 128, SP_ / 128);
    gemm_wmma_kernel<true><<<gProj, 256, 0, stream>>>(hs16, DIM_, SP_, wq16, DIM_, 0, 1, 0,
                                                      bq, q32, DIM_, DIM_, 0);
    gemm_wmma_kernel<true><<<gProj, 256, 0, stream>>>(hs16, DIM_, SP_, wk16, DIM_, 0, 1, 0,
                                                      bk, k32, DIM_, DIM_, 0);
    gemm_wmma_kernel<true><<<gProj, 256, 0, stream>>>(hs16, DIM_, SP_, wv16, DIM_, 0, 1, 0,
                                                      bv, v32, DIM_, DIM_, 0);

    // LayerNorm + RoPE + layout change
    ln_rope_kernel<<<SP_ * HEADS_, 32, 0, stream>>>(q32, k32, v32, cosb, sinb,
                                                    nqw, nqb, nkw, nkb,
                                                    q16, k16, vt16);

    // Attention
    attn_wmma_kernel<<<dim3(SP_ / 64, HEADS_), 128, 0, stream>>>(q16, k16, vt16, attn32);

    // Conv branch: conv1 (GELU, f16 out) then conv2 (f32 out), 3 taps, shift 256
    gemm_wmma_kernel<false><<<dim3(RANK_ / 128, SV_ / 128), 256, 0, stream>>>(
        hs16 + (size_t)T_ * DIM_, DIM_, SV_, w1t16, DIM_, RANK_ * DIM_, 3, 256,
        nullptr, c1o16, RANK_, DIM_, 1);
    gemm_wmma_kernel<false><<<dim3(DIM_ / 128, SV_ / 128), 256, 0, stream>>>(
        c1o16, RANK_, SV_, w2t16, RANK_, DIM_ * RANK_, 3, 256,
        nullptr, cbr32, DIM_, RANK_, 0);

    // attn + conv_branch -> f16 projection input
    combine_kernel<<<(nHS + 255) / 256, 256, 0, stream>>>(attn32, cbr32, pin16);

    // Output projection with permuted store: d_out = [rows T..S) ++ [rows 0..T)
    gemm_wmma_kernel<true><<<gProj, 256, 0, stream>>>(pin16, DIM_, SP_, wo16, DIM_, 0, 1, 0,
                                                      bo, out, DIM_, DIM_, 2);
    (void)in_sizes; (void)n_in; (void)out_size; (void)ws_size;
}
